// DecoderBlock_71124658422385
// MI455X (gfx1250) — compile-verified, matrix operand order checked by reading, not run on hardware
//
#include <hip/hip_runtime.h>
#include <math.h>

// ---------------------------------------------------------------- types
typedef __bf16 bf16_t;
typedef __attribute__((ext_vector_type(16))) __bf16 v16bf;
typedef __attribute__((ext_vector_type(8)))  float  v8f;

union FragB {
    v16bf  v;
    uint4  q[2];
    bf16_t e[16];
};

__device__ __forceinline__ bf16_t f2bf(float f) {
    unsigned int u = __float_as_uint(f);
    unsigned int r = u + 0x7FFFu + ((u >> 16) & 1u);   // round to nearest even
    unsigned short s = (unsigned short)(r >> 16);
    return __builtin_bit_cast(bf16_t, s);
}

__device__ __forceinline__ v8f wmma_bf16(const v16bf& a, const v16bf& b, const v8f& c) {
    return __builtin_amdgcn_wmma_f32_16x16x32_bf16(false, a, false, b, (short)0, c, false, false);
}

__device__ __forceinline__ v8f v8f_zero() {
    v8f z;
#pragma unroll
    for (int i = 0; i < 8; ++i) z[i] = 0.0f;
    return z;
}

// ---------------------------------------------------------------- constants
#define DMODEL 1024
#define NHEAD  16
#define HDIM   64
#define FFDIM  4096
#define BATCH  4
#define NTOK   1024
#define MROWS  (BATCH * NTOK)   // 4096

// ---------------------------------------------------------------- fp32 -> bf16 (straight)
__global__ void cvt_f32_to_bf16(const float* __restrict__ in, bf16_t* __restrict__ out, int n) {
    int i = blockIdx.x * blockDim.x + threadIdx.x;
    if (i < n) out[i] = f2bf(in[i]);
}

// ---------------------------------------------------------------- fp32 [K][N] -> bf16 [N][K]
__global__ __launch_bounds__(256)
void cvt_transpose_bf16(const float* __restrict__ in, bf16_t* __restrict__ out,
                        int Kdim, int Ndim) {
    __shared__ float tile[32][33];
    const int k0 = blockIdx.y * 32;
    const int n0 = blockIdx.x * 32;
    const int tx = threadIdx.x & 31;
    const int ty = threadIdx.x >> 5;   // 0..7
#pragma unroll
    for (int i = ty; i < 32; i += 8)
        tile[i][tx] = in[(size_t)(k0 + i) * Ndim + n0 + tx];
    __syncthreads();
#pragma unroll
    for (int i = ty; i < 32; i += 8)
        out[(size_t)(n0 + i) * Kdim + k0 + tx] = f2bf(tile[tx][i]);
}

// ---------------------------------------------------------------- GEMM (bf16 WMMA, fp32 acc)
// C[M,N] = A[M,K] @ B[K,N], with B supplied TRANSPOSED as Bt[N][K] bf16.
// flags: bit0 = bf16 output, bit1 = exact GELU, bit2 = write output transposed [N][M].
#define TM 128
#define TN 128
#define TK 32

__global__ __launch_bounds__(256)
void gemm_bf16_wmma(const bf16_t* __restrict__ A, const bf16_t* __restrict__ Bt,
                    void* __restrict__ Cout, const float* __restrict__ bias,
                    int M, int N, int K, int flags) {
    __shared__ __align__(16) bf16_t sA[2][TM * TK];   // [row][k]  2 x 8 KB
    __shared__ __align__(16) bf16_t sB[2][TN * TK];   // [n][k]    2 x 8 KB

    const int t    = threadIdx.x;
    const int lane = t & 31;
    const int wave = t >> 5;                 // 0..7
    const int wrow = (wave & 3) * 32;        // 0,32,64,96
    const int wcol = (wave >> 2) * 64;       // 0,64
    const int m0   = blockIdx.y * TM;
    const int n0   = blockIdx.x * TN;
    const int hi   = lane >> 4;
    const int l16  = lane & 15;
    const int kb   = hi * 8;

    // staging: thread covers one 16-elem half-row of the 32-wide k slab
    const int srow = t >> 1;                 // 0..127
    const int sseg = (t & 1) * 16;           // 0 or 16
    const bf16_t* gA = A  + (size_t)(m0 + srow) * K + sseg;
    const bf16_t* gB = Bt + (size_t)(n0 + srow) * K + sseg;

    v8f acc[2][4];
#pragma unroll
    for (int i = 0; i < 2; ++i)
#pragma unroll
        for (int j = 0; j < 4; ++j) acc[i][j] = v8f_zero();

    const int nk = K / TK;
    uint4 pa0, pa1, pb0, pb1;

    // prologue: stage slab 0 into buffer 0
    pa0 = *reinterpret_cast<const uint4*>(gA);
    pa1 = *reinterpret_cast<const uint4*>(gA + 8);
    pb0 = *reinterpret_cast<const uint4*>(gB);
    pb1 = *reinterpret_cast<const uint4*>(gB + 8);
    {
        uint4* dA = reinterpret_cast<uint4*>(&sA[0][srow * TK + sseg]);
        dA[0] = pa0; dA[1] = pa1;
        uint4* dB = reinterpret_cast<uint4*>(&sB[0][srow * TK + sseg]);
        dB[0] = pb0; dB[1] = pb1;
    }
    __syncthreads();

    for (int kt = 0; kt < nk; ++kt) {
        const int cur = kt & 1;
        const int nxt = cur ^ 1;
        const bool more = (kt + 1) < nk;
        if (more) {
            int ko = (kt + 1) * TK;
            pa0 = *reinterpret_cast<const uint4*>(gA + ko);
            pa1 = *reinterpret_cast<const uint4*>(gA + ko + 8);
            pb0 = *reinterpret_cast<const uint4*>(gB + ko);
            pb1 = *reinterpret_cast<const uint4*>(gB + ko + 8);
        }

        FragB af[2], bfr[4];
#pragma unroll
        for (int i = 0; i < 2; ++i) {
            int row = wrow + i * 16 + l16;
            af[i].q[0] = *reinterpret_cast<const uint4*>(&sA[cur][row * TK + kb]);
            af[i].q[1] = *reinterpret_cast<const uint4*>(&sA[cur][row * TK + kb + 16]);
        }
#pragma unroll
        for (int j = 0; j < 4; ++j) {
            int col = wcol + j * 16 + l16;
            bfr[j].q[0] = *reinterpret_cast<const uint4*>(&sB[cur][col * TK + kb]);
            bfr[j].q[1] = *reinterpret_cast<const uint4*>(&sB[cur][col * TK + kb + 16]);
        }
#pragma unroll
        for (int i = 0; i < 2; ++i)
#pragma unroll
            for (int j = 0; j < 4; ++j)
                acc[i][j] = wmma_bf16(af[i].v, bfr[j].v, acc[i][j]);

        if (more) {
            uint4* dA = reinterpret_cast<uint4*>(&sA[nxt][srow * TK + sseg]);
            dA[0] = pa0; dA[1] = pa1;
            uint4* dB = reinterpret_cast<uint4*>(&sB[nxt][srow * TK + sseg]);
            dB[0] = pb0; dB[1] = pb1;
        }
        __syncthreads();
    }

    const bool out_bf   = (flags & 1) != 0;
    const bool do_gelu  = (flags & 2) != 0;
    const bool out_tr   = (flags & 4) != 0;
#pragma unroll
    for (int i = 0; i < 2; ++i) {
#pragma unroll
        for (int j = 0; j < 4; ++j) {
            int col = n0 + wcol + j * 16 + l16;
            float bv = bias ? bias[col] : 0.0f;
#pragma unroll
            for (int r = 0; r < 8; ++r) {
                int row = m0 + wrow + i * 16 + hi * 8 + r;
                float v = acc[i][j][r] + bv;
                if (do_gelu) v = 0.5f * v * (1.0f + erff(v * 0.70710678118654752440f));
                if (out_tr)       ((bf16_t*)Cout)[(size_t)col * M + row] = f2bf(v);
                else if (out_bf)  ((bf16_t*)Cout)[(size_t)row * N + col] = f2bf(v);
                else              ((float*) Cout)[(size_t)row * N + col] = v;
            }
        }
    }
}

// ---------------------------------------------------------------- flash attention
// One wave per (batch, head, 16-query tile).
// Q,K bf16 [b*Nq + tok][DMODEL]; V supplied TRANSPOSED: Vt bf16 [DMODEL][Mtot]. O fp32.
__global__ __launch_bounds__(32)
void attn_wmma(const bf16_t* __restrict__ Qg, const bf16_t* __restrict__ Kg,
               const bf16_t* __restrict__ Vt, float* __restrict__ Og,
               int Nq, int Nk, int Mtot, int causal, float scale) {
    __shared__ __align__(16) bf16_t sP[16 * 32];   // P re-layout staging, 1 KB

    const int lane = threadIdx.x & 31;
    const int hi   = lane >> 4;
    const int l16  = lane & 15;
    const int kb   = hi * 8;
    const int qt   = blockIdx.x;
    const int h    = blockIdx.y;
    const int b    = blockIdx.z;
    const int qbase = qt * 16;
    const size_t qoff = (size_t)b * Nq * DMODEL;
    const size_t koff = (size_t)b * Nk * DMODEL;

    // Q fragments: 16 queries x 64 head-dims = two 16x32 A fragments
    FragB qa[2];
#pragma unroll
    for (int kc = 0; kc < 2; ++kc) {
        const bf16_t* base = Qg + qoff + (size_t)(qbase + l16) * DMODEL + h * HDIM + kc * 32 + kb;
        qa[kc].q[0] = *reinterpret_cast<const uint4*>(base);
        qa[kc].q[1] = *reinterpret_cast<const uint4*>(base + 16);
    }

    v8f o[4];
    float m[8], lsum[8];
#pragma unroll
    for (int n = 0; n < 4; ++n) o[n] = v8f_zero();
#pragma unroll
    for (int r = 0; r < 8; ++r) { m[r] = -1e30f; lsum[r] = 0.0f; }

    int kend = Nk;
    if (causal) {
        kend = (qbase + 16 + 31) & ~31;
        if (kend > Nk) kend = Nk;
    }

    for (int j0 = 0; j0 < kend; j0 += 32) {
        // ---- S = Q K^T for 32 keys (two 16x16 C tiles)
        v8f s[2];
#pragma unroll
        for (int st = 0; st < 2; ++st) {
            FragB kb0, kb1;   // B = K^T: lane holds key column, elems walk head-dim
            const bf16_t* base = Kg + koff + (size_t)(j0 + st * 16 + l16) * DMODEL + h * HDIM + kb;
            kb0.q[0] = *reinterpret_cast<const uint4*>(base);
            kb0.q[1] = *reinterpret_cast<const uint4*>(base + 16);
            kb1.q[0] = *reinterpret_cast<const uint4*>(base + 32);
            kb1.q[1] = *reinterpret_cast<const uint4*>(base + 48);
            v8f z = v8f_zero();
            z = wmma_bf16(qa[0].v, kb0.v, z);
            z = wmma_bf16(qa[1].v, kb1.v, z);
            s[st] = z;
        }
        // ---- scale + causal mask
#pragma unroll
        for (int st = 0; st < 2; ++st) {
            int col = j0 + st * 16 + l16;
#pragma unroll
            for (int r = 0; r < 8; ++r) {
                int row = qbase + r + hi * 8;
                float v = s[st][r] * scale;
                if (causal && col > row) v = -1e30f;
                s[st][r] = v;
            }
        }
        // ---- online softmax (row reductions across 16-lane halves)
        float alpha[8];
#pragma unroll
        for (int r = 0; r < 8; ++r) {
            float rm = fmaxf(s[0][r], s[1][r]);
#pragma unroll
            for (int off = 1; off < 16; off <<= 1)
                rm = fmaxf(rm, __shfl_xor(rm, off, 32));
            float mn = fmaxf(m[r], rm);
            alpha[r] = __expf(m[r] - mn);
            m[r] = mn;
            float p0 = __expf(s[0][r] - mn);
            float p1 = __expf(s[1][r] - mn);
            s[0][r] = p0; s[1][r] = p1;
            float rs = p0 + p1;
#pragma unroll
            for (int off = 1; off < 16; off <<= 1)
                rs += __shfl_xor(rs, off, 32);
            lsum[r] = lsum[r] * alpha[r] + rs;
        }
#pragma unroll
        for (int n = 0; n < 4; ++n)
#pragma unroll
            for (int r = 0; r < 8; ++r) o[n][r] *= alpha[r];

        // ---- C-layout P -> A-layout bf16 fragment via LDS (single wave, LDS in-order)
#pragma unroll
        for (int r = 0; r < 8; ++r) {
            int row = r + hi * 8;
            sP[row * 32 + l16]      = f2bf(s[0][r]);
            sP[row * 32 + 16 + l16] = f2bf(s[1][r]);
        }
        FragB pa;
        pa.q[0] = *reinterpret_cast<const uint4*>(&sP[l16 * 32 + kb]);
        pa.q[1] = *reinterpret_cast<const uint4*>(&sP[l16 * 32 + kb + 16]);

        // ---- O += P (16x32) * V (32x64): V fragments straight from Vt, vectorized
#pragma unroll
        for (int n = 0; n < 4; ++n) {
            FragB vb;
            const bf16_t* vbase = Vt + (size_t)(h * HDIM + n * 16 + l16) * Mtot
                                     + (size_t)b * Nk + j0 + kb;
            vb.q[0] = *reinterpret_cast<const uint4*>(vbase);
            vb.q[1] = *reinterpret_cast<const uint4*>(vbase + 16);
            o[n] = wmma_bf16(pa.v, vb.v, o[n]);
        }
    }

    // ---- normalize + store
#pragma unroll
    for (int n = 0; n < 4; ++n)
#pragma unroll
        for (int r = 0; r < 8; ++r) {
            int row = qbase + r + hi * 8;
            Og[qoff + (size_t)row * DMODEL + h * HDIM + n * 16 + l16] = o[n][r] / lsum[r];
        }
}

// ---------------------------------------------------------------- residual + LayerNorm
__global__ __launch_bounds__(256)
void add_ln(const float* __restrict__ a, const float* __restrict__ resid,
            const float* __restrict__ g, const float* __restrict__ beta,
            float* __restrict__ yf, bf16_t* __restrict__ ybf) {
    __shared__ float red[16];
    const int row = blockIdx.x;
    const int t = threadIdx.x;
    const float* pa = a     + (size_t)row * DMODEL;
    const float* pr = resid + (size_t)row * DMODEL;

    float x[4], s = 0.0f, ss = 0.0f;
#pragma unroll
    for (int i = 0; i < 4; ++i) {
        int c = t + i * 256;
        float v = pa[c] + pr[c];
        x[i] = v; s += v; ss += v * v;
    }
    for (int off = 1; off < 32; off <<= 1) {
        s  += __shfl_xor(s,  off, 32);
        ss += __shfl_xor(ss, off, 32);
    }
    int wave = t >> 5, lane = t & 31;
    if (lane == 0) { red[wave] = s; red[8 + wave] = ss; }
    __syncthreads();
    if (wave == 0) {
        float s2  = (lane < 8) ? red[lane]     : 0.0f;
        float ss2 = (lane < 8) ? red[8 + lane] : 0.0f;
        for (int off = 1; off < 8; off <<= 1) {
            s2  += __shfl_xor(s2,  off, 32);
            ss2 += __shfl_xor(ss2, off, 32);
        }
        if (lane == 0) { red[0] = s2; red[1] = ss2; }
    }
    __syncthreads();
    const float mean = red[0] * (1.0f / DMODEL);
    const float var  = red[1] * (1.0f / DMODEL) - mean * mean;
    const float rstd = rsqrtf(var + 1e-5f);
#pragma unroll
    for (int i = 0; i < 4; ++i) {
        int c = t + i * 256;
        float v = (x[i] - mean) * rstd * g[c] + beta[c];
        yf[(size_t)row * DMODEL + c] = v;
        if (ybf) ybf[(size_t)row * DMODEL + c] = f2bf(v);
    }
}

// ---------------------------------------------------------------- launch
extern "C" void kernel_launch(void* const* d_in, const int* in_sizes, int n_in,
                              void* d_out, int out_size, void* d_ws, size_t ws_size,
                              hipStream_t stream) {
    const float* x    = (const float*)d_in[0];
    const float* enc  = (const float*)d_in[1];
    const float* Wq1  = (const float*)d_in[2];
    const float* Wk1  = (const float*)d_in[3];
    const float* Wv1  = (const float*)d_in[4];
    const float* g1   = (const float*)d_in[5];
    const float* b1   = (const float*)d_in[6];
    const float* Wq2  = (const float*)d_in[7];
    const float* Wk2  = (const float*)d_in[8];
    const float* Wv2  = (const float*)d_in[9];
    const float* g2   = (const float*)d_in[10];
    const float* b2   = (const float*)d_in[11];
    const float* Wff1 = (const float*)d_in[12];
    const float* bff1 = (const float*)d_in[13];
    const float* Wff2 = (const float*)d_in[14];
    const float* bff2 = (const float*)d_in[15];
    const float* g3   = (const float*)d_in[16];
    const float* b3   = (const float*)d_in[17];

    char* ws = (char*)d_ws;
    const size_t MB = 1024u * 1024u;
    bf16_t* wq1b  = (bf16_t*)(ws + 0   * MB);   // all weights stored transposed [N][K]
    bf16_t* wk1b  = (bf16_t*)(ws + 2   * MB);
    bf16_t* wv1b  = (bf16_t*)(ws + 4   * MB);
    bf16_t* wq2b  = (bf16_t*)(ws + 6   * MB);
    bf16_t* wk2b  = (bf16_t*)(ws + 8   * MB);
    bf16_t* wv2b  = (bf16_t*)(ws + 10  * MB);
    bf16_t* wff1b = (bf16_t*)(ws + 12  * MB);
    bf16_t* wff2b = (bf16_t*)(ws + 20  * MB);
    bf16_t* xbf   = (bf16_t*)(ws + 28  * MB);   // activation bf16 (reused: x, y1, y2)
    bf16_t* encbf = (bf16_t*)(ws + 36  * MB);
    bf16_t* qbf   = (bf16_t*)(ws + 44  * MB);
    bf16_t* kbf   = (bf16_t*)(ws + 52  * MB);
    bf16_t* vbf   = (bf16_t*)(ws + 60  * MB);   // V stored transposed [DMODEL][MROWS]
    float*  ctx   = (float*) (ws + 68  * MB);   // attn ctx / ff2 out
    float*  yA    = (float*) (ws + 84  * MB);
    float*  yB    = (float*) (ws + 100 * MB);
    bf16_t* hbf   = (bf16_t*)(ws + 116 * MB);   // ends at 148 MB

    auto cvt = [&](const float* src, bf16_t* dst, int n) {
        cvt_f32_to_bf16<<<dim3((n + 255) / 256), dim3(256), 0, stream>>>(src, dst, n);
    };
    auto cvtT = [&](const float* src, bf16_t* dst, int Kd, int Nd) {
        cvt_transpose_bf16<<<dim3(Nd / 32, Kd / 32), dim3(256), 0, stream>>>(src, dst, Kd, Nd);
    };
    auto gemm = [&](const bf16_t* A, const bf16_t* Bt, void* C, const float* bias,
                    int M, int N, int K, int flags) {
        gemm_bf16_wmma<<<dim3(N / TN, M / TM), dim3(256), 0, stream>>>(A, Bt, C, bias, M, N, K, flags);
    };

    // weights -> bf16, transposed [N][K]
    cvtT(Wq1,  wq1b,  DMODEL, DMODEL);
    cvtT(Wk1,  wk1b,  DMODEL, DMODEL);
    cvtT(Wv1,  wv1b,  DMODEL, DMODEL);
    cvtT(Wq2,  wq2b,  DMODEL, DMODEL);
    cvtT(Wk2,  wk2b,  DMODEL, DMODEL);
    cvtT(Wv2,  wv2b,  DMODEL, DMODEL);
    cvtT(Wff1, wff1b, DMODEL, FFDIM);
    cvtT(Wff2, wff2b, FFDIM,  DMODEL);
    cvt(x,   xbf,   MROWS * DMODEL);
    cvt(enc, encbf, MROWS * DMODEL);

    const float scale = 0.03125f;   // 1/sqrt(DMODEL), as in reference
    dim3 agrid(NTOK / 16, NHEAD, BATCH);

    // ---- block 1: causal self-attention
    gemm(xbf, wq1b, qbf, nullptr, MROWS, DMODEL, DMODEL, 1);
    gemm(xbf, wk1b, kbf, nullptr, MROWS, DMODEL, DMODEL, 1);
    gemm(xbf, wv1b, vbf, nullptr, MROWS, DMODEL, DMODEL, 1 | 4);   // transposed output
    attn_wmma<<<agrid, 32, 0, stream>>>(qbf, kbf, vbf, ctx, NTOK, NTOK, MROWS, 1, scale);
    add_ln<<<MROWS, 256, 0, stream>>>(ctx, x, g1, b1, yA, xbf);    // xbf := y1 (bf16)

    // ---- block 2: cross-attention over encoder output
    gemm(xbf,   wq2b, qbf, nullptr, MROWS, DMODEL, DMODEL, 1);
    gemm(encbf, wk2b, kbf, nullptr, MROWS, DMODEL, DMODEL, 1);
    gemm(encbf, wv2b, vbf, nullptr, MROWS, DMODEL, DMODEL, 1 | 4); // transposed output
    attn_wmma<<<agrid, 32, 0, stream>>>(qbf, kbf, vbf, ctx, NTOK, NTOK, MROWS, 0, scale);
    add_ln<<<MROWS, 256, 0, stream>>>(ctx, yA, g2, b2, yB, xbf);   // xbf := y2 (bf16)

    // ---- block 3: FFN (bias+GELU fused), residual, final LN
    gemm(xbf, wff1b, hbf, bff1, MROWS, FFDIM, DMODEL, 1 | 2);      // bf16 out + GELU
    gemm(hbf, wff2b, ctx, bff2, MROWS, DMODEL, FFDIM, 0);          // fp32 out
    add_ln<<<MROWS, 256, 0, stream>>>(ctx, yB, g3, b3, (float*)d_out, nullptr);
}
